// Encoder_47330539602344
// MI455X (gfx1250) — compile-verified, matrix operand order checked by reading, not run on hardware
//
#include <hip/hip_runtime.h>
#include <hip/hip_bf16.h>

// Problem constants (from reference)
#define BB 1024
#define TT 128
#define VV 64
#define EE 256
#define HH 512
#define GG 2048   // 4*H

// Padded LDS row stride for the h tile: 520 bf16 = 1040 B (4-bank rotation/row)
#define HLDS_STRIDE 520
#define MTILE 32

typedef __attribute__((ext_vector_type(16))) __bf16 bf16x16;
typedef __attribute__((ext_vector_type(8)))  float  f32x8;

union FragU {
    uint4   u[2];
    bf16x16 v;
};

// Gate tile (after K loop) aliases the h tile (dead after K loop).
union SMem {
    float          g[MTILE * 512];          // 64 KB gate tile
    __hip_bfloat16 h[MTILE * HLDS_STRIDE];  // 32.5 KB staged h tile
};

// ---------------------------------------------------------------------------
// Kernel: sequence lengths (first EOS(0) position + 1, else T)
// ---------------------------------------------------------------------------
__global__ __launch_bounds__(256) void k_lengths(const int* __restrict__ ids,
                                                 int* __restrict__ len) {
    int b = blockIdx.x * blockDim.x + threadIdx.x;
    if (b >= BB) return;
    const int* row = ids + (size_t)b * TT;
    int L = TT;
    for (int t = 0; t < TT; ++t) {
        if (row[t] == 0) { L = t + 1; break; }
    }
    len[b] = L;
}

// ---------------------------------------------------------------------------
// Kernel: emb_g[v, n] = sum_e emb[v,e] * W_ih[n,e] + b_ih[n] + b_hh[n]
// Tiny (64 x 2048 x 256) one-shot precompute; fp32 for fidelity.
// ---------------------------------------------------------------------------
__global__ __launch_bounds__(256) void k_embg(const float* __restrict__ emb,
                                              const float* __restrict__ W_ih,
                                              const float* __restrict__ b_ih,
                                              const float* __restrict__ b_hh,
                                              float* __restrict__ emb_g) {
    int v = blockIdx.x;                       // 0..63
    int n = blockIdx.y * 256 + threadIdx.x;   // 0..2047
    const float* er = emb  + (size_t)v * EE;
    const float* wr = W_ih + (size_t)n * EE;
    float s = 0.f;
#pragma unroll 4
    for (int e = 0; e < EE; ++e) s += er[e] * wr[e];
    emb_g[(size_t)v * GG + n] = s + b_ih[n] + b_hh[n];
}

// ---------------------------------------------------------------------------
// Kernel: cast W_hh (f32, [4H,H] row-major, K-contiguous) -> bf16
// ---------------------------------------------------------------------------
__global__ __launch_bounds__(256) void k_cast_whh(const float* __restrict__ src,
                                                  __hip_bfloat16* __restrict__ dst) {
    size_t i = (size_t)blockIdx.x * 256 + threadIdx.x;
    if (i < (size_t)GG * HH) dst[i] = __float2bfloat16(src[i]);
}

// ---------------------------------------------------------------------------
// Kernel: zero-init h_state (f32), c_state (f32), h_bf16 ping buffer
// ---------------------------------------------------------------------------
__global__ __launch_bounds__(256) void k_init_state(float* __restrict__ h_state,
                                                    float* __restrict__ c_state,
                                                    __hip_bfloat16* __restrict__ hbf) {
    size_t i = (size_t)blockIdx.x * 256 + threadIdx.x;
    if (i < (size_t)BB * HH) {
        h_state[i] = 0.f;
        c_state[i] = 0.f;
        hbf[i]     = __float2bfloat16(0.f);
    }
}

__device__ __forceinline__ float sigf(float x) {
    return 1.0f / (1.0f + __expf(-x));
}

// ---------------------------------------------------------------------------
// Per-timestep fused kernel:
//   gates = emb_g[token] + h_{t-1} @ W_hh^T   (bf16 WMMA, f32 accum)
//   LSTM cell + dropout mask + length masking, writes hs / h / c.
//
// Grid: (B/32, H/128), 256 threads (8 wave32s).
// Block computes a 32(batch) x 512(virtual-col) gate tile; virtual col
// vc = gate*128 + jj maps to real col n = gate*512 + j0 + jj, so the epilogue
// has all four gates of each hidden unit locally (via LDS).
//
// Data movement:
//   - 32x512 bf16 h tile staged once per block into LDS with
//     global_load_async_to_lds_b128 (ASYNCcnt path, no VGPR transit),
//     padded rows for conflict-free fragment reads.
//   - W_hh fragments (per-wave distinct) double-buffered in registers and
//     reused across BOTH M sub-tiles: every L2 B-fragment feeds 2 WMMAs,
//     halving per-step L2 weight traffic (64 MB/step) => compute-bound.
// ---------------------------------------------------------------------------
__global__ __launch_bounds__(256) void k_step(
    const __hip_bfloat16* __restrict__ hbf_in,   // [B,H] bf16 (prev h)
    __hip_bfloat16* __restrict__ hbf_out,        // [B,H] bf16 (new h)
    const __hip_bfloat16* __restrict__ whh,      // [4H,H] bf16 row-major
    const float* __restrict__ emb_g,             // [V,4H]
    const int* __restrict__ ids,                 // [B,T]
    const int* __restrict__ len,                 // [B]
    const float* __restrict__ mask,              // [B,H]
    float* __restrict__ hs,                      // [B,T,H]
    float* __restrict__ h_state,                 // [B,H] f32 (carried / last_h)
    float* __restrict__ c_state,                 // [B,H] f32 (carried / last_c)
    int t) {

    __shared__ SMem smem;

    const int m0   = blockIdx.x * MTILE;         // batch base
    const int j0   = blockIdx.y * 128;           // hidden-slice base
    const int wave = threadIdx.x >> 5;
    const int lane = threadIdx.x & 31;
    const int half = lane >> 4;                  // K sub-block select
    const int lm   = lane & 15;                  // M (A) / N (B) within tile

    // ---- Stage h tile (32 rows x 512 bf16 = 2048 x 16B chunks) into LDS ----
    // Async global->LDS: each lane copies 16B directly into LDS (ASYNCcnt).
    {
        int c = threadIdx.x;
#pragma unroll
        for (int it = 0; it < 8; ++it, c += 256) {
            int row = c >> 6;                    // 0..31
            int col = c & 63;                    // 16B chunk within row
            const __hip_bfloat16* gp = hbf_in + (size_t)(m0 + row) * HH + col * 8;
            unsigned loff = (unsigned)(uintptr_t)&smem.h[row * HLDS_STRIDE + col * 8];
            asm volatile("global_load_async_to_lds_b128 %0, %1, off"
                         :: "v"(loff), "v"(gp)
                         : "memory");
        }
        asm volatile("s_wait_asynccnt 0x0" ::: "memory");
    }
    __syncthreads();

    // B fragment source rows: W_hh[n, :] for the 4 owned 16-col groups
    const __hip_bfloat16* brow[4];
    int ncol[4];
#pragma unroll
    for (int tn = 0; tn < 4; ++tn) {
        int vc = wave * 64 + tn * 16 + lm;       // virtual column
        int g  = vc >> 7;                        // gate index
        int jj = vc & 127;
        int n  = g * HH + j0 + jj;               // real gate column
        ncol[tn] = vc - lm;                      // tile base vcol (for LDS store)
        brow[tn] = whh + (size_t)n * HH + half * 8;
    }

    f32x8 acc[2][4] = {};

    // Prime the B double-buffer for k = 0
    FragU bcur[4];
#pragma unroll
    for (int tn = 0; tn < 4; ++tn) {
        const __hip_bfloat16* p = brow[tn];
        bcur[tn].u[0] = *(const uint4*)(p);
        bcur[tn].u[1] = *(const uint4*)(p + 16);
    }

    const __hip_bfloat16* als0 = &smem.h[(lm)      * HLDS_STRIDE + half * 8];
    const __hip_bfloat16* als1 = &smem.h[(16 + lm) * HLDS_STRIDE + half * 8];

    // K loop over H=512 in steps of 32 (bf16 WMMA K), B frags pipelined,
    // each B fragment reused by both M sub-tiles.
#pragma unroll
    for (int k = 0; k < HH; k += 32) {
        FragU bnxt[4];
        if (k + 32 < HH) {
#pragma unroll
            for (int tn = 0; tn < 4; ++tn) {
                const __hip_bfloat16* p = brow[tn] + (k + 32);
                bnxt[tn].u[0] = *(const uint4*)(p);
                bnxt[tn].u[1] = *(const uint4*)(p + 16);
            }
        }

        FragU a0, a1;
        a0.u[0] = *(const uint4*)(als0 + k);
        a0.u[1] = *(const uint4*)(als0 + k + 16);
        a1.u[0] = *(const uint4*)(als1 + k);
        a1.u[1] = *(const uint4*)(als1 + k + 16);

#pragma unroll
        for (int tn = 0; tn < 4; ++tn) {
            acc[0][tn] = __builtin_amdgcn_wmma_f32_16x16x32_bf16(
                false, a0.v, false, bcur[tn].v, (short)0, acc[0][tn], false, false);
        }
#pragma unroll
        for (int tn = 0; tn < 4; ++tn) {
            acc[1][tn] = __builtin_amdgcn_wmma_f32_16x16x32_bf16(
                false, a1.v, false, bcur[tn].v, (short)0, acc[1][tn], false, false);
        }

        if (k + 32 < HH) {
#pragma unroll
            for (int tn = 0; tn < 4; ++tn) bcur[tn] = bnxt[tn];
        }
    }

    // h tile is dead; gate tile aliases it. Barrier before overwriting.
    __syncthreads();

    // Spill accumulators: element (m = ms*16 + r + 8*half, vc = tilebase + lm)
#pragma unroll
    for (int ms = 0; ms < 2; ++ms) {
#pragma unroll
        for (int tn = 0; tn < 4; ++tn) {
#pragma unroll
            for (int r = 0; r < 8; ++r) {
                smem.g[(ms * 16 + r + 8 * half) * 512 + ncol[tn] + lm] = acc[ms][tn][r];
            }
        }
    }
    __syncthreads();

    // Epilogue: 32 rows x 128 hidden units = 4096 elements, 16 per thread
#pragma unroll
    for (int e = 0; e < 16; ++e) {
        int idx = threadIdx.x + e * 256;
        int m   = idx >> 7;          // 0..31
        int jj  = idx & 127;         // 0..127
        int b   = m0 + m;
        int j   = j0 + jj;

        int tok = ids[(size_t)b * TT + t];
        const float* eg = emb_g + (size_t)tok * GG;

        float gi = smem.g[m * 512 +   0 + jj] + eg[0 * HH + j];
        float gf = smem.g[m * 512 + 128 + jj] + eg[1 * HH + j];
        float gg = smem.g[m * 512 + 256 + jj] + eg[2 * HH + j];
        float go = smem.g[m * 512 + 384 + jj] + eg[3 * HH + j];

        float iv = sigf(gi);
        float fv = sigf(gf);
        float gv = tanhf(gg);
        float ov = sigf(go);

        size_t sidx = (size_t)b * HH + j;
        float cprev = c_state[sidx];
        float hprev = h_state[sidx];

        float cn = fv * cprev + iv * gv;
        float hn = ov * tanhf(cn) * mask[sidx];

        bool act = t < len[b];
        float hout = act ? hn : hprev;
        float cout = act ? cn : cprev;

        hs[((size_t)b * TT + t) * HH + j] = act ? hn : 0.f;
        h_state[sidx] = hout;
        c_state[sidx] = cout;
        hbf_out[sidx] = __float2bfloat16(hout);
    }
}

// ---------------------------------------------------------------------------
// Host-side launch
// ---------------------------------------------------------------------------
extern "C" void kernel_launch(void* const* d_in, const int* in_sizes, int n_in,
                              void* d_out, int out_size, void* d_ws, size_t ws_size,
                              hipStream_t stream) {
    const int*   ids  = (const int*)d_in[0];
    const float* emb  = (const float*)d_in[1];
    const float* W_ih = (const float*)d_in[2];
    const float* W_hh = (const float*)d_in[3];
    const float* b_ih = (const float*)d_in[4];
    const float* b_hh = (const float*)d_in[5];
    const float* mask = (const float*)d_in[6];

    float* out     = (float*)d_out;
    float* hs      = out;                                   // [B,T,H]
    float* h_state = out + (size_t)BB * TT * HH;            // [B,H] -> last_h
    float* c_state = h_state + (size_t)BB * HH;             // [B,H] -> last_c

    // Workspace layout (all 16B aligned)
    char* ws = (char*)d_ws;
    float* emb_g = (float*)ws;                 ws += (size_t)VV * GG * sizeof(float);
    __hip_bfloat16* whh_bf = (__hip_bfloat16*)ws; ws += (size_t)GG * HH * sizeof(__hip_bfloat16);
    __hip_bfloat16* hbf0 = (__hip_bfloat16*)ws;  ws += (size_t)BB * HH * sizeof(__hip_bfloat16);
    __hip_bfloat16* hbf1 = (__hip_bfloat16*)ws;  ws += (size_t)BB * HH * sizeof(__hip_bfloat16);
    int* len = (int*)ws;

    // One-shot precomputation
    k_lengths<<<(BB + 255) / 256, 256, 0, stream>>>(ids, len);
    k_embg<<<dim3(VV, GG / 256), 256, 0, stream>>>(emb, W_ih, b_ih, b_hh, emb_g);
    k_cast_whh<<<(int)(((size_t)GG * HH + 255) / 256), 256, 0, stream>>>(W_hh, whh_bf);
    k_init_state<<<(int)(((size_t)BB * HH + 255) / 256), 256, 0, stream>>>(h_state, c_state, hbf0);

    // Sequential recurrence: one fused WMMA GEMM + cell kernel per timestep,
    // bf16 h feed ping-ponged to avoid cross-block RAW within a step.
    dim3 grid(BB / MTILE, HH / 128);
    for (int t = 0; t < TT; ++t) {
        const __hip_bfloat16* src = (t & 1) ? hbf1 : hbf0;
        __hip_bfloat16*       dst = (t & 1) ? hbf0 : hbf1;
        k_step<<<grid, 256, 0, stream>>>(src, dst, whh_bf, emb_g, ids, len, mask,
                                         hs, h_state, c_state, t);
    }
}
